// ISTFT_69982197121745
// MI455X (gfx1250) — matrix operand order repaired and checked by reading, not compiled, not run on hardware
//
#include <hip/hip_runtime.h>
#include <stdint.h>

// CDNA5 (gfx1250) WMMA types
typedef __attribute__((ext_vector_type(16))) __bf16 v16bf;
typedef __attribute__((ext_vector_type(8)))  float  v8f;

#define FFT_BIN   513
#define NCH       1026          // 2*FFT_BIN channels after concat
#define NFRAMES   2048
#define KW        1024          // kernel width (N_FFT)
#define HOPSZ     256
#define OUTLEN    524032        // per-batch output samples
#define BM        128           // f0 rows per workgroup
#define KCH       8             // channels per reduction chunk -> K = 4*KCH = 32
#define NCHUNK    129           // ceil(1026/8)
#define NTHR      1024          // 32 waves
#define LROW      40            // padded LDS row (32 bf16 + 8 pad) = 80B, 16B aligned

#define KPACK_BYTES   (NCH * KW * 4)             // 4,202,496
#define BFMT_ELEMS    (NCHUNK * 256 * 32)        // 1,056,768
#define BCHUNK_BYTES  (256 * 32 * 2)             // 16,384 per chunk per hi/lo

// Async global->LDS path (CDNA5): gate on builtin availability; fallback = VGPR staging.
#if defined(__AMDGCN__) && __has_builtin(__builtin_amdgcn_global_load_async_to_lds_b128)
#define HAVE_ASYNC_LDS 1
// Parameter type per hipcc diagnostic: pointer to int __vector(4), generic AS.
typedef __attribute__((__vector_size__(4 * sizeof(int)))) int v4i_ld;
#else
#define HAVE_ASYNC_LDS 0
#endif

__device__ __forceinline__ void wait_async0() {
#if HAVE_ASYNC_LDS
#if __has_builtin(__builtin_amdgcn_s_wait_asynccnt)
  __builtin_amdgcn_s_wait_asynccnt(0);
#else
  asm volatile("s_wait_asynccnt 0x0" ::: "memory");
#endif
#endif
}

__device__ __forceinline__ uint32_t bf16_rne(float f) {
  uint32_t u = __float_as_uint(f);
  return (u + 0x7FFFu + ((u >> 16) & 1u)) >> 16;
}
__device__ __forceinline__ float bf16f(uint32_t h) { return __uint_as_float(h << 16); }

// Pass 0a: pack kernel matrix as (bf16_hi << 16) | bf16_lo  (fallback path operand).
__global__ void pack_bf16x2(const float* __restrict__ src, uint32_t* __restrict__ dst, int n) {
  int i = blockIdx.x * blockDim.x + threadIdx.x;
  if (i >= n) return;
  float f = src[i];
  uint32_t h = bf16_rne(f);
  uint32_t l = bf16_rne(f - bf16f(h));
  dst[i] = (h << 16) | (l & 0xFFFFu);
}

// Pass 0b: pre-format B operand in final LDS tile order, split hi/lo:
//   bfmt[cc][q][r] = kernel[cc*8 + r/4][(r%4)*256 + q], zero-padded past c=1025.
__global__ void pack_bfmt(const float* __restrict__ kern, uint16_t* __restrict__ bh,
                          uint16_t* __restrict__ bl, int total) {
  int i = blockIdx.x * blockDim.x + threadIdx.x;
  if (i >= total) return;
  const int r  = i & 31;
  const int q  = (i >> 5) & 255;
  const int cc = i >> 13;                 // 256*32 = 8192 elements per chunk
  const int c  = cc * KCH + (r >> 2);
  const int k  = (r & 3) * HOPSZ + q;
  const float f = (c < NCH) ? kern[(size_t)c * KW + k] : 0.0f;
  const uint32_t h = bf16_rne(f);
  const uint32_t l = bf16_rne(f - bf16f(h));
  bh[i] = (uint16_t)h;
  bl[i] = (uint16_t)l;
}

union AFrag { v16bf v; uint4 q[2]; };

struct Tile {  // one pipeline stage of LDS operands (61,440 B); two stages = 120 KB
  uint16_t At_hi[BM][LROW];
  uint16_t At_lo[BM][LROW];
  uint16_t Bt_hi[256][LROW];
  uint16_t Bt_lo[256][LROW];
};

// out[b, 256*f0 + q - 512] = sum_{j=0..3} sum_c x[b,c,f0-j] * kernel[c, 256*j + q]
// One dense GEMM: rows (b,f0), cols q, reduction r = 4*c + j (length 4104, zero-padded).
__global__ __launch_bounds__(NTHR)
void istft_wmma_bf16x3(const float* __restrict__ realp, const float* __restrict__ imagp,
                       const uint32_t* __restrict__ kpack,
                       const uint16_t* __restrict__ bfmt_hi,
                       const uint16_t* __restrict__ bfmt_lo,
                       float* __restrict__ out) {
  __shared__ __align__(16) Tile tiles[2];

  const int tid    = threadIdx.x;
  const int lane   = tid & 31;
  const int lane16 = lane & 15;
  const int hi16   = lane >> 4;            // 0: lanes 0-15, 1: lanes 16-31
  const int wid    = tid >> 5;             // 0..31
  const int wave_m = wid >> 2;             // 0..7  -> M offset = wave_m*16
  const int wave_n = wid & 3;              // 0..3  -> N offset = wave_n*64
  const int b      = blockIdx.y;
  const int F0     = 2 + BM * (int)blockIdx.x;   // f0 base, valid f0 in [2, 2048]

  v8f acc[4];
#pragma unroll
  for (int nt = 0; nt < 4; ++nt)
#pragma unroll
    for (int v = 0; v < 8; ++v) acc[nt][v] = 0.0f;

  // ---- staging maps (per thread, loop invariant) ----
  const int sm   = tid & (BM - 1);   // A-tile row (f0 - F0)
  const int scl  = tid >> 7;         // A-tile channel-in-chunk 0..7
  const int sq   = tid & 255;        // B-tile q row (fallback path)
  const int sgrp = tid >> 8;         // 0..3 -> handles cl = 2*sgrp + {0,1} (fallback path)

  // A-side frame window for this thread: frames f0-3..f0, fixed for all chunks.
  const int f0me = F0 + sm;
  int  fr[4];
  bool fv[4];
#pragma unroll
  for (int i = 0; i < 4; ++i) {
    fr[i] = f0me - 3 + i;
    fv[i] = (fr[i] >= 0) && (fr[i] < NFRAMES);
  }

  // ---- A operand: global loads for chunk C0 into registers ----
  auto load_A = [&](int C0, float* xv) {
    const int c   = C0 + scl;
    const bool cv = (c < NCH);
    const float* row = realp;  // only dereferenced when cv
    if (cv) {
      row = (c < FFT_BIN) ? (realp + (size_t)(b * FFT_BIN + c) * NFRAMES)
                          : (imagp + (size_t)(b * FFT_BIN + (c - FFT_BIN)) * NFRAMES);
    }
#pragma unroll
    for (int i = 0; i < 4; ++i) xv[i] = (cv && fv[i]) ? row[fr[i]] : 0.0f;
  };

  // ---- A operand: split fp32 -> bf16 hi/lo, store to LDS ----
  auto store_A = [&](Tile* T, const float* xv) {
    uint64_t hv = 0, lv = 0;
#pragma unroll
    for (int j = 0; j < 4; ++j) {       // At[m][4*cl + j] = x[f0 - j] = xv[3-j]
      const float f = xv[3 - j];
      const uint32_t hh = bf16_rne(f);
      const uint32_t ll = bf16_rne(f - bf16f(hh));
      hv |= (uint64_t)(hh & 0xFFFFu) << (16 * j);
      lv |= (uint64_t)(ll & 0xFFFFu) << (16 * j);
    }
    *reinterpret_cast<uint64_t*>(&T->At_hi[sm][4 * scl]) = hv;
    *reinterpret_cast<uint64_t*>(&T->At_lo[sm][4 * scl]) = lv;
  };

#if HAVE_ASYNC_LDS
  // ---- B operand: async global->LDS, 2x b128 per thread, tracked by ASYNCcnt ----
  const int brow = tid >> 2;            // Bt row q (0..255)
  const int bseg = (tid & 3) * 8;       // 16B segment within the 64B row payload
  auto async_B = [&](Tile* T, int cc) {
    const size_t goff = (size_t)cc * BCHUNK_BYTES + (size_t)tid * 16;
    __builtin_amdgcn_global_load_async_to_lds_b128(
        (v4i_ld*)((const char*)bfmt_hi + goff),
        (v4i_ld*)&T->Bt_hi[brow][bseg], 0, 0);
    __builtin_amdgcn_global_load_async_to_lds_b128(
        (v4i_ld*)((const char*)bfmt_lo + goff),
        (v4i_ld*)&T->Bt_lo[brow][bseg], 0, 0);
  };
#else
  // ---- B operand (fallback): packed kernel via VGPRs ----
  auto load_B = [&](int C0, uint32_t* kp) {
#pragma unroll
    for (int ci = 0; ci < 2; ++ci) {
      const int c2 = C0 + sgrp * 2 + ci;
      const size_t base = (size_t)c2 * KW + sq;
#pragma unroll
      for (int j = 0; j < 4; ++j)
        kp[ci * 4 + j] = (c2 < NCH) ? kpack[base + j * HOPSZ] : 0u;
    }
  };
  auto store_B = [&](Tile* T, const uint32_t* kp) {
#pragma unroll
    for (int ci = 0; ci < 2; ++ci) {
      const int cl = sgrp * 2 + ci;
      uint64_t bh = 0, bl = 0;
#pragma unroll
      for (int j = 0; j < 4; ++j) {
        const uint32_t p = kp[ci * 4 + j];
        bh |= (uint64_t)(p >> 16)     << (16 * j);
        bl |= (uint64_t)(p & 0xFFFFu) << (16 * j);
      }
      *reinterpret_cast<uint64_t*>(&T->Bt_hi[sq][4 * cl]) = bh;
      *reinterpret_cast<uint64_t*>(&T->Bt_lo[sq][4 * cl]) = bl;
    }
  };
#endif

  // ---- compute stage: 12 WMMAs on one LDS tile ----
  auto compute = [&](const Tile* T) {
    // A fragments (16x32 bf16): lanes<16 hold K 0-7 & 16-23, lanes>=16 hold K 8-15 & 24-31
    AFrag ahi, alo;
    const int m  = wave_m * 16 + lane16;
    const int ka = hi16 ? 8 : 0;
    ahi.q[0] = *reinterpret_cast<const uint4*>(&T->At_hi[m][ka]);
    ahi.q[1] = *reinterpret_cast<const uint4*>(&T->At_hi[m][ka + 16]);
    alo.q[0] = *reinterpret_cast<const uint4*>(&T->At_lo[m][ka]);
    alo.q[1] = *reinterpret_cast<const uint4*>(&T->At_lo[m][ka + 16]);
    // B fragments (32x16 bf16): lanes<16 hold K 0-7 & 16-23, lanes>=16 hold K 8-15 & 24-31
#pragma unroll
    for (int nt = 0; nt < 4; ++nt) {
      const int n  = wave_n * 64 + nt * 16 + lane16;
      const int kb = hi16 ? 8 : 0;
      AFrag bhi, blo;
      bhi.q[0] = *reinterpret_cast<const uint4*>(&T->Bt_hi[n][kb]);
      bhi.q[1] = *reinterpret_cast<const uint4*>(&T->Bt_hi[n][kb + 16]);
      blo.q[0] = *reinterpret_cast<const uint4*>(&T->Bt_lo[n][kb]);
      blo.q[1] = *reinterpret_cast<const uint4*>(&T->Bt_lo[n][kb + 16]);
      // bf16x3: hi*hi + hi*lo + lo*hi, fp32 accumulate (lo*lo dropped, ~2^-18 rel)
      acc[nt] = __builtin_amdgcn_wmma_f32_16x16x32_bf16(false, ahi.v, false, bhi.v,
                                                        (short)0, acc[nt], false, false);
      acc[nt] = __builtin_amdgcn_wmma_f32_16x16x32_bf16(false, ahi.v, false, blo.v,
                                                        (short)0, acc[nt], false, false);
      acc[nt] = __builtin_amdgcn_wmma_f32_16x16x32_bf16(false, alo.v, false, bhi.v,
                                                        (short)0, acc[nt], false, false);
    }
  };

  // ---- software-pipelined main loop: double-buffered LDS, 1 barrier/iteration ----
  float xv[4];
#if !HAVE_ASYNC_LDS
  uint32_t kp[8];
#endif

  load_A(0, xv);
#if HAVE_ASYNC_LDS
  async_B(&tiles[0], 0);
#else
  load_B(0, kp);
#endif
  store_A(&tiles[0], xv);
#if !HAVE_ASYNC_LDS
  store_B(&tiles[0], kp);
#endif
  wait_async0();
  __syncthreads();

#pragma unroll 1
  for (int cc = 0; cc < NCHUNK; ++cc) {
    const bool havenext = (cc + 1 < NCHUNK);
    if (havenext) {                       // loads/copies for cc+1 in flight...
      load_A((cc + 1) * KCH, xv);
#if HAVE_ASYNC_LDS
      async_B(&tiles[(cc + 1) & 1], cc + 1);
#else
      load_B((cc + 1) * KCH, kp);
#endif
    }
    compute(&tiles[cc & 1]);              // ...hidden behind 12 WMMAs
    if (havenext) {
      store_A(&tiles[(cc + 1) & 1], xv);
#if !HAVE_ASYNC_LDS
      store_B(&tiles[(cc + 1) & 1], kp);
#endif
    }
    wait_async0();
    __syncthreads();
  }

  // ---- epilogue: C layout = VGPR v -> M = v + 8*hi16; lane -> N = lane16 ----
  const int qbase = wave_n * 64 + lane16;
#pragma unroll
  for (int nt = 0; nt < 4; ++nt) {
#pragma unroll
    for (int v = 0; v < 8; ++v) {
      const int m  = wave_m * 16 + v + hi16 * 8;
      const int f0 = F0 + m;
      if (f0 <= NFRAMES) {  // valid f0 in [2, 2048]; each output written exactly once
        const size_t t = (size_t)(f0 - 2) * 256 + (qbase + nt * 16);
        out[(size_t)b * OUTLEN + t] = acc[nt][v];
      }
    }
  }
}

extern "C" void kernel_launch(void* const* d_in, const int* in_sizes, int n_in,
                              void* d_out, int out_size, void* d_ws, size_t ws_size,
                              hipStream_t stream) {
  const float* realp = (const float*)d_in[0];   // (16, 513, 2048) fp32
  const float* imagp = (const float*)d_in[1];   // (16, 513, 2048) fp32
  const float* kern  = (const float*)d_in[2];   // (1026, 1024) fp32
  float* outp = (float*)d_out;                  // (16, 1, 524032) fp32

  // Workspace layout (~8.4 MB): kpack | bfmt_hi | bfmt_lo. Both formats are always
  // built so the host launch sequence is independent of the device-side path choice.
  uint32_t* kpack = (uint32_t*)d_ws;
  uint16_t* bh = (uint16_t*)((char*)d_ws + KPACK_BYTES);
  uint16_t* bl = bh + BFMT_ELEMS;

  const int nk = in_sizes[2];                   // 1026*1024
  pack_bf16x2<<<dim3((nk + 255) / 256), dim3(256), 0, stream>>>(kern, kpack, nk);
  pack_bfmt<<<dim3((BFMT_ELEMS + 255) / 256), dim3(256), 0, stream>>>(kern, bh, bl, BFMT_ELEMS);

  // 16 f0-tiles (BM=128 over f0 in [2,2049], guarded) x 16 batches
  istft_wmma_bf16x3<<<dim3(16, 16), dim3(NTHR), 0, stream>>>(realp, imagp, kpack, bh, bl, outp);
}
